// PAIR_MODULE_43155831390393
// MI455X (gfx1250) — matrix operand order repaired and checked by reading, not compiled, not run on hardware
//
#include <hip/hip_runtime.h>

typedef __attribute__((ext_vector_type(16))) __bf16 v16bf;
typedef __attribute__((ext_vector_type(8)))  __bf16 v8bf;
typedef __attribute__((ext_vector_type(4)))  __bf16 v4bf;
typedef __attribute__((ext_vector_type(8)))  float  v8f;

#define LDIM 1024
#define KDIM 2048
#define BROWS 1280   // B*R = 128*10
#define RR 10
#define NK (KDIM / 32)   // 64 K-steps
#define LP 40            // padded LDS row stride (bf16): 80 B, 16B-aligned, conflict-free

// ws layout (floats): [0..2047] v ; [2048] c ; [2064..3343] sh ; [3344..4623] sq

// ---------------- Kernel 1: v = mlp_w1 @ mlp_w2, c = mlp_b1.mlp_w2 + mlp_b2, zero sh/sq
__global__ void __launch_bounds__(256)
prep_kernel(const float* __restrict__ w1, const float* __restrict__ b1,
            const float* __restrict__ w2, const float* __restrict__ b2,
            float* __restrict__ ws) {
  __shared__ float red[256];
  const int blk = blockIdx.x;
  const int t = threadIdx.x;
  if (blk < 2049) {
    const float* rowp = (blk < 2048) ? (w1 + (size_t)blk * LDIM) : b1;
    float acc = 0.f;
    #pragma unroll
    for (int it = 0; it < 4; ++it) {
      int k = t + it * 256;
      acc += rowp[k] * w2[k];
    }
    red[t] = acc;
    __syncthreads();
    for (int s = 128; s > 0; s >>= 1) {
      if (t < s) red[t] += red[t + s];
      __syncthreads();
    }
    if (t == 0) {
      if (blk < 2048) ws[blk] = red[0];
      else            ws[2048] = red[0] + b2[0];
    }
  } else {
    int idx = (blk - 2049) * 256 + t;
    if (idx < 2 * BROWS) ws[2064 + idx] = 0.f;
  }
}

// ---------------- Kernel 2: fused gated-embedding GEMM + fold onto v ----------------
// Block tile 128(M) x 64(N), BK=32. grid.x = (1280/128)*(1024/64) = 160, grid.y = 2.
// 256 threads = 8 waves laid out 4(M) x 2(N); each wave: 2x2 subtiles of 16x16,
// two accumulators (E and G) per subtile -> 8 WMMAs per wave per K-step.
__global__ void __launch_bounds__(256)
emb_score_kernel(const float* __restrict__ hist, const float* __restrict__ ques,
                 const float* __restrict__ h_wy, const float* __restrict__ h_by,
                 const float* __restrict__ h_wg, const float* __restrict__ h_bg,
                 const float* __restrict__ q_wy, const float* __restrict__ q_by,
                 const float* __restrict__ q_wg, const float* __restrict__ q_bg,
                 float* __restrict__ ws) {
  const int path = blockIdx.y;
  const float* __restrict__ X    = path ? ques : hist;
  const float* __restrict__ Wy   = path ? q_wy : h_wy;
  const float* __restrict__ Wg   = path ? q_wg : h_wg;
  const float* __restrict__ By   = path ? q_by : h_by;
  const float* __restrict__ Bg   = path ? q_bg : h_bg;
  const float* __restrict__ vseg = ws + (path ? LDIM : 0);
  float* __restrict__ sout       = ws + 2064 + (path ? BROWS : 0);

  const int tileId = blockIdx.x;
  const int mBase = (tileId >> 4) * 128;   // 16 N-tiles per M-tile
  const int nBase = (tileId & 15) * 64;

  __shared__ __align__(16) __bf16 sA [2][128][LP];  // [m][k]
  __shared__ __align__(16) __bf16 sWy[2][64][LP];   // [n][k] (transposed on store)
  __shared__ __align__(16) __bf16 sWg[2][64][LP];

  const int tid    = threadIdx.x;
  const int lane   = tid & 31;
  const int wave   = tid >> 5;
  const int waveM  = wave >> 1;  // 0..3 -> 32-row band
  const int waveN  = wave & 1;   // 0..1 -> 32-col band
  const int half   = lane >> 4;  // 0 or 1
  const int lrow   = lane & 15;

  // --- staging coordinates ---
  const int ar = tid >> 3;             // A rows: ar, ar+32, ar+64, ar+96
  const int ac = (tid & 7) * 4;        // A col (float4)
  const int wk = tid >> 4;             // W k: wk, wk+16
  const int wn = (tid & 15) * 4;       // W n (float4)

  const float* Ap  = X  + (size_t)(mBase + ar) * KDIM + ac;
  const float* Wyp = Wy + (size_t)wk * LDIM + nBase + wn;
  const float* Wgp = Wg + (size_t)wk * LDIM + nBase + wn;

  v8f accE[2][2] = {};   // [mi][ni]
  v8f accG[2][2] = {};

  float4 aR[4], yR[2], gR[2];

  auto load_regs = [&](int k) {
    const size_t kb = (size_t)k * 32;
    #pragma unroll
    for (int i = 0; i < 4; ++i)
      aR[i] = *(const float4*)(Ap + (size_t)(32 * i) * KDIM + kb);
    #pragma unroll
    for (int i = 0; i < 2; ++i) {
      yR[i] = *(const float4*)(Wyp + (kb + 16 * i) * LDIM);
      gR[i] = *(const float4*)(Wgp + (kb + 16 * i) * LDIM);
    }
    if (k + 2 < NK) {  // prefetch streamed X two tiles ahead (global_prefetch_b8)
      __builtin_prefetch(Ap + (size_t)(k + 2) * 32, 0, 0);
      __builtin_prefetch(Ap + (size_t)64 * KDIM + (size_t)(k + 2) * 32, 0, 0);
    }
  };

  auto store_tile = [&](int buf) {
    #pragma unroll
    for (int i = 0; i < 4; ++i) {
      v4bf a;
      a[0] = (__bf16)aR[i].x; a[1] = (__bf16)aR[i].y;
      a[2] = (__bf16)aR[i].z; a[3] = (__bf16)aR[i].w;
      *(v4bf*)&sA[buf][ar + 32 * i][ac] = a;
    }
    #pragma unroll
    for (int i = 0; i < 2; ++i) {
      const int kk = wk + 16 * i;
      sWy[buf][wn + 0][kk] = (__bf16)yR[i].x;
      sWy[buf][wn + 1][kk] = (__bf16)yR[i].y;
      sWy[buf][wn + 2][kk] = (__bf16)yR[i].z;
      sWy[buf][wn + 3][kk] = (__bf16)yR[i].w;
      sWg[buf][wn + 0][kk] = (__bf16)gR[i].x;
      sWg[buf][wn + 1][kk] = (__bf16)gR[i].y;
      sWg[buf][wn + 2][kk] = (__bf16)gR[i].z;
      sWg[buf][wn + 3][kk] = (__bf16)gR[i].w;
    }
  };

  auto compute_tile = [&](int buf) {
    v16bf afrag[2], bY[2], bG[2];
    #pragma unroll
    for (int mi = 0; mi < 2; ++mi) {
      // A (16-bit 16x32): lanes 0-15 hold K {0-15}, lanes 16-31 K {16-31}
      const int am = waveM * 32 + mi * 16 + lrow;
      v8bf lo = *(const v8bf*)&sA[buf][am][half * 16];
      v8bf hi = *(const v8bf*)&sA[buf][am][half * 16 + 8];
      #pragma unroll
      for (int e = 0; e < 8; ++e) { afrag[mi][e] = lo[e]; afrag[mi][e + 8] = hi[e]; }
    }
    #pragma unroll
    for (int ni = 0; ni < 2; ++ni) {
      const int bn = waveN * 32 + ni * 16 + lrow;
      v8bf ylo = *(const v8bf*)&sWy[buf][bn][half * 16];
      v8bf yhi = *(const v8bf*)&sWy[buf][bn][half * 16 + 8];
      v8bf glo = *(const v8bf*)&sWg[buf][bn][half * 16];
      v8bf ghi = *(const v8bf*)&sWg[buf][bn][half * 16 + 8];
      #pragma unroll
      for (int e = 0; e < 8; ++e) {
        bY[ni][e] = ylo[e]; bY[ni][e + 8] = yhi[e];
        bG[ni][e] = glo[e]; bG[ni][e + 8] = ghi[e];
      }
    }
    #pragma unroll
    for (int mi = 0; mi < 2; ++mi)
      #pragma unroll
      for (int ni = 0; ni < 2; ++ni) {
        accE[mi][ni] = __builtin_amdgcn_wmma_f32_16x16x32_bf16(
            false, afrag[mi], false, bY[ni], (short)0, accE[mi][ni], false, false);
        accG[mi][ni] = __builtin_amdgcn_wmma_f32_16x16x32_bf16(
            false, afrag[mi], false, bG[ni], (short)0, accG[mi][ni], false, false);
      }
  };

  // ---- software-pipelined main loop (double-buffered LDS) ----
  load_regs(0);
  store_tile(0);
  for (int k = 0; k < NK - 1; ++k) {
    __syncthreads();                 // buf[k&1] ready for all waves
    load_regs(k + 1);                // issue next tile's global loads early
    compute_tile(k & 1);             // 8 WMMAs on current buffer
    store_tile((k + 1) & 1);         // fill the other buffer
  }
  __syncthreads();
  compute_tile((NK - 1) & 1);

  // ---- Epilogue: bias + tanh*sigmoid, fold onto v, reduce over N, atomic into s ----
  const int nG0 = nBase + waveN * 32 + lrow;      // C layout: N = lrow
  const int nG1 = nG0 + 16;
  const float by0 = By[nG0], bg0 = Bg[nG0], vv0 = vseg[nG0];
  const float by1 = By[nG1], bg1 = Bg[nG1], vv1 = vseg[nG1];

  #pragma unroll
  for (int mi = 0; mi < 2; ++mi) {
    float pr[8];
    #pragma unroll
    for (int r = 0; r < 8; ++r) {
      float e0 = accE[mi][0][r] + by0, g0 = accG[mi][0][r] + bg0;
      float e1 = accE[mi][1][r] + by1, g1 = accG[mi][1][r] + bg1;
      pr[r] = tanhf(e0) * (1.f / (1.f + expf(-g0))) * vv0 +
              tanhf(e1) * (1.f / (1.f + expf(-g1))) * vv1;
    }
    #pragma unroll
    for (int r = 0; r < 8; ++r) {
      float p = pr[r];
      p += __shfl_xor(p, 1, 32);
      p += __shfl_xor(p, 2, 32);
      p += __shfl_xor(p, 4, 32);
      p += __shfl_xor(p, 8, 32);    // stays within each 16-lane half
      pr[r] = p;
    }
    if (lrow == 0) {
      // C layout: VGPR r holds M = r (lanes 0-15) or r+8 (lanes 16-31)
      const int rowBase = mBase + waveM * 32 + mi * 16 + half * 8;
      #pragma unroll
      for (int r = 0; r < 8; ++r)
        atomicAdd(&sout[rowBase + r], pr[r]);
    }
  }
}

// ---------------- Kernel 3: causal Gumbel-softmax, one wave per (b,i) row ----------
__global__ void __launch_bounds__(32)
softmax_kernel(const float* __restrict__ gumbel,
               const float* __restrict__ att_w, const float* __restrict__ att_b,
               const float* __restrict__ ws, float* __restrict__ out) {
  const int bi = blockIdx.x;           // 0..1279 = b*R + i
  const int b = bi / RR, i = bi % RR;
  const int j = threadIdx.x;           // lane
  const float* sh = ws + 2064;
  const float* sq = sh + BROWS;
  const float c  = ws[2048];
  const float w0 = att_w[0], w1 = att_w[1], b0 = att_b[0];

  const bool valid = (j < RR) && (j <= i);
  float y = -1e30f;
  if (valid) {
    float score = sh[b * RR + j] + sq[b * RR + i] + c;
    float delta = (float)(i - j + 1);
    float u = gumbel[(size_t)bi * RR + j];
    float g = -logf(-logf(u + 1e-20f) + 1e-20f);
    y = score * w0 + delta * w1 + b0 + g;   // TAU = 1
  }
  float m = y;
  for (int mask = 16; mask; mask >>= 1) m = fmaxf(m, __shfl_xor(m, mask, 32));
  float e = valid ? expf(y - m) : 0.f;
  float s = e;
  for (int mask = 16; mask; mask >>= 1) s += __shfl_xor(s, mask, 32);
  if (j < RR) out[(size_t)bi * RR + j] = valid ? (e / s) : 0.f;
}

extern "C" void kernel_launch(void* const* d_in, const int* in_sizes, int n_in,
                              void* d_out, int out_size, void* d_ws, size_t ws_size,
                              hipStream_t stream) {
  const float* hist   = (const float*)d_in[0];
  const float* ques   = (const float*)d_in[1];
  const float* gumbel = (const float*)d_in[2];
  const float* h_wy = (const float*)d_in[3];
  const float* h_by = (const float*)d_in[4];
  const float* h_wg = (const float*)d_in[5];
  const float* h_bg = (const float*)d_in[6];
  const float* q_wy = (const float*)d_in[7];
  const float* q_by = (const float*)d_in[8];
  const float* q_wg = (const float*)d_in[9];
  const float* q_bg = (const float*)d_in[10];
  const float* mlp_w1 = (const float*)d_in[11];
  const float* mlp_b1 = (const float*)d_in[12];
  const float* mlp_w2 = (const float*)d_in[13];
  const float* mlp_b2 = (const float*)d_in[14];
  const float* att_w  = (const float*)d_in[15];
  const float* att_b  = (const float*)d_in[16];
  float* ws  = (float*)d_ws;
  float* out = (float*)d_out;

  prep_kernel<<<2059, 256, 0, stream>>>(mlp_w1, mlp_b1, mlp_w2, mlp_b2, ws);

  dim3 grid(160, 2);
  emb_score_kernel<<<grid, 256, 0, stream>>>(hist, ques,
                                             h_wy, h_by, h_wg, h_bg,
                                             q_wy, q_by, q_wg, q_bg, ws);

  softmax_kernel<<<1280, 32, 0, stream>>>(gumbel, att_w, att_b, ws, out);
}
